// LinearAttention_43181601194830
// MI455X (gfx1250) — matrix operand order repaired and checked by reading, not compiled, not run on hardware
//
#include <hip/hip_runtime.h>
#include <hip/hip_bf16.h>
#include <stdint.h>

#define B_   4
#define S_   4096
#define HID_ 1024
#define H_   16
#define D_   64
#define BS_  (B_ * S_)

typedef __attribute__((ext_vector_type(16))) __bf16 v16bf;
typedef __attribute__((ext_vector_type(8)))  __bf16 v8bf;
typedef __attribute__((ext_vector_type(8)))  float  v8f;

// ---------------------------------------------------------------------------
// WMMA helpers (CDNA5 16x16x32 bf16, f32 accumulate)
// ---------------------------------------------------------------------------
static __device__ __forceinline__ v16bf pack16(v8bf lo, v8bf hi) {
  v16bf a;
#pragma unroll
  for (int i = 0; i < 8; ++i) { a[i] = lo[i]; a[i + 8] = hi[i]; }
  return a;
}

// A-fragment (16x32 bf16) from a row-major array with leading dim `ld`.
static __device__ __forceinline__ v16bf afrag(const __bf16* base, int ld,
                                              int row0, int kk, int lane) {
  int r  = row0 + (lane & 15);
  int kb = kk + ((lane >> 4) << 4);
  const __bf16* p = base + (size_t)r * ld + kb;
  v8bf lo = *(const v8bf*)(p);
  v8bf hi = *(const v8bf*)(p + 8);
  return pack16(lo, hi);
}

// B-fragment (32x16 bf16): lane holds column col0+(L&15); 16 contiguous K
// elems starting at kk + (L>=16 ? 16 : 0).  Source layout: [col][k] row-major.
static __device__ __forceinline__ v16bf bfrag(const __bf16* base, int ld,
                                              int col0, int kk, int lane) {
  int c  = col0 + (lane & 15);
  int kb = kk + ((lane >> 4) << 4);
  const __bf16* p = base + (size_t)c * ld + kb;
  v8bf lo = *(const v8bf*)(p);
  v8bf hi = *(const v8bf*)(p + 8);
  return pack16(lo, hi);
}

// Same as bfrag but converting from an f32 source (state matrix in LDS).
static __device__ __forceinline__ v16bf bfrag_f32(const float* base, int ld,
                                                  int col0, int kk, int lane) {
  int c  = col0 + (lane & 15);
  int kb = kk + ((lane >> 4) << 4);
  const float* p = base + (size_t)c * ld + kb;
  v16bf b;
#pragma unroll
  for (int i = 0; i < 16; ++i) b[i] = (__bf16)p[i];
  return b;
}

static __device__ __forceinline__ v8f wmma_bf16(v16bf a, v16bf b, v8f c) {
  return __builtin_amdgcn_wmma_f32_16x16x32_bf16(false, a, false, b,
                                                 (short)0, c, false, false);
}

// ---------------------------------------------------------------------------
// CDNA5 async Global->LDS copy (16 bytes per lane), tracked by ASYNCcnt.
// LDS aperture keeps the byte offset in the low 32 bits of the generic addr.
// ---------------------------------------------------------------------------
static __device__ __forceinline__ void async_copy16(void* lds_dst,
                                                    const void* gsrc) {
  unsigned           l = (unsigned)(uintptr_t)lds_dst;
  unsigned long long g = (unsigned long long)(uintptr_t)gsrc;
  asm volatile("global_load_async_to_lds_b128 %0, %1, off"
               :: "v"(l), "v"(g) : "memory");
}
template <int N>
static __device__ __forceinline__ void async_wait() {
  asm volatile("s_wait_asynccnt %0" :: "i"(N) : "memory");
}

// ---------------------------------------------------------------------------
// Prep kernels
// ---------------------------------------------------------------------------
__global__ void cvt_f32_bf16_kernel(const float* __restrict__ src,
                                    __bf16* __restrict__ dst, int n) {
  int i = blockIdx.x * blockDim.x + threadIdx.x;
  int stride = gridDim.x * blockDim.x;
  for (; i < n; i += stride) dst[i] = (__bf16)src[i];
}

// WT[n][k] = (bf16) W[k][n], LDS-tiled so both global sides are coalesced.
__global__ void __launch_bounds__(256)
transpose_cvt_kernel(const float* __restrict__ W, __bf16* __restrict__ WT,
                     int K, int N) {
  __shared__ float tile[32][33];
  int nb = blockIdx.x * 32;
  int kb = blockIdx.y * 32;
  int tx = threadIdx.x & 31;
  int ty = threadIdx.x >> 5;  // 0..7
#pragma unroll
  for (int i = ty; i < 32; i += 8)
    tile[i][tx] = W[(size_t)(kb + i) * N + nb + tx];
  __syncthreads();
#pragma unroll
  for (int i = ty; i < 32; i += 8)
    WT[(size_t)(nb + i) * K + kb + tx] = (__bf16)tile[tx][i];
}

// ---------------------------------------------------------------------------
// GEMM: C[M,N] = A[M,K](bf16) @ BT[N,K](bf16)^T, +bias, optional elu+1.
// 256 threads (8 waves), tile 128x128, K-step 32, double-buffered async
// Global->LDS staging of the B tile (s_wait_asynccnt 2 overlaps DMA/WMMA).
// NSTORE: write row-major Cout (bf16 or f32).  TSTORE: also write a
// head-transposed bf16 copy CoutT[b][h][e][s] (one 16B store per tile/lane).
// ---------------------------------------------------------------------------
template <int PHI, int OUTBF16, int NSTORE, int TSTORE>
__global__ void __launch_bounds__(256)
gemm_bf16_kernel(const __bf16* __restrict__ A, const __bf16* __restrict__ BT,
                 const float* __restrict__ bias, void* __restrict__ Cout,
                 __bf16* __restrict__ CoutT, int M, int N, int K) {
  __shared__ alignas(16) __bf16 Bs[2][128 * 32];  // 2 x 8 KB

  const int tid  = threadIdx.x;
  const int lane = tid & 31;
  const int wave = tid >> 5;        // 0..7
  const int n0 = blockIdx.x * 128;
  const int m0 = blockIdx.y * 128;
  const int rowb = m0 + wave * 16;  // 16-row M strip per wave

  v8f acc[8] = {};

  // prologue: stage tile 0
#pragma unroll
  for (int c = tid; c < 512; c += 256) {
    int row = c >> 2, part = c & 3;
    async_copy16(Bs[0] + row * 32 + part * 8,
                 BT + (size_t)(n0 + row) * K + part * 8);
  }

  for (int kk = 0; kk < K; kk += 32) {
    const int cur = (kk >> 5) & 1;
    if (kk + 32 < K) {  // prefetch next tile into other buffer
#pragma unroll
      for (int c = tid; c < 512; c += 256) {
        int row = c >> 2, part = c & 3;
        async_copy16(Bs[cur ^ 1] + row * 32 + part * 8,
                     BT + (size_t)(n0 + row) * K + kk + 32 + part * 8);
      }
      async_wait<2>();   // in-order: <=2 outstanding => current tile resident
    } else {
      async_wait<0>();
    }
    __syncthreads();

    v16bf af = afrag(A, K, rowb, kk, lane);
#pragma unroll
    for (int nt = 0; nt < 8; ++nt) {
      v16bf bf_ = bfrag(Bs[cur], 32, nt * 16, 0, lane);
      acc[nt] = wmma_bf16(af, bf_, acc[nt]);
    }
    __syncthreads();
  }

  // epilogue: row = rowb + r + 8*(lane>=16), col = n0 + nt*16 + (lane&15)
  const int ro = (lane >> 4) << 3;
  const int cL = lane & 15;
  const int row0 = rowb + ro;          // 8 contiguous output rows per lane
#pragma unroll
  for (int nt = 0; nt < 8; ++nt) {
    int col = n0 + nt * 16 + cL;
    float bv = bias ? bias[col] : 0.0f;
    float val[8];
#pragma unroll
    for (int r = 0; r < 8; ++r) {
      float v = acc[nt][r] + bv;
      if (PHI) v = (v > 0.0f) ? (v + 1.0f) : __expf(v);  // elu(v)+1
      val[r] = v;
    }
    if (NSTORE) {
#pragma unroll
      for (int r = 0; r < 8; ++r) {
        if (OUTBF16)
          ((__bf16*)Cout)[(size_t)(row0 + r) * N + col] = (__bf16)val[r];
        else
          ((float*)Cout)[(size_t)(row0 + r) * N + col] = val[r];
      }
    }
    if (TSTORE) {
      // CoutT[b][h][e][s]: 8 contiguous s values -> single 16B store
      int bidx = row0 >> 12;           // row0 / S_
      int s0   = row0 & (S_ - 1);
      int hh   = col >> 6;
      int e    = col & 63;
      v8bf tp;
#pragma unroll
      for (int r = 0; r < 8; ++r) tp[r] = (__bf16)val[r];
      *(v8bf*)(CoutT + (((size_t)bidx * H_ + hh) * 64 + e) * S_ + s0) = tp;
    }
  }
}

// ---------------------------------------------------------------------------
// Chunked linear-attention scan.  One block per (b,h); 128 threads (4 waves).
// K^T and V^T chunks come pre-transposed from global (written by the
// projection GEMM epilogue), so there are no LDS transposes here.
// State in LDS:  Mst[e][d] = KV^T (f32), KsS[d] = sum of k (f32).
// Per chunk (C=64):
//   S   = Q @ K^T            (WMMA, causal-masked -> Ss bf16)
//   den = Q . KsS + rowsum(Ss)
//   num = Ss @ V + Q @ Mst   (WMMA)
//   Mst += V^T @ K           (WMMA)
//   KsS += colsum(K)         (contiguous via K^T)
// ---------------------------------------------------------------------------
__global__ void __launch_bounds__(128)
linattn_scan_kernel(const __bf16* __restrict__ Qg, const __bf16* __restrict__ Kg,
                    const __bf16* __restrict__ KTg, const __bf16* __restrict__ VTg,
                    __bf16* __restrict__ Og) {
  __shared__ alignas(16) float  Mst[64 * 64];     // state KV^T  [e][d]
  __shared__ alignas(16) float  KsS[64];          // state sum(k)
  __shared__ alignas(16) __bf16 Qs[64 * 64];      // [i][d]
  __shared__ alignas(16) __bf16 Ks[64 * 64];      // [j][d]
  __shared__ alignas(16) __bf16 KTs[64 * 64];     // [d][j]
  __shared__ alignas(16) __bf16 VTs[64 * 64];     // [e][j]
  __shared__ alignas(16) __bf16 Ss[64 * 64];      // masked scores [i][j]
  __shared__ float den[64];

  const int tid  = threadIdx.x;
  const int lane = tid & 31;
  const int wave = tid >> 5;
  const int bh = blockIdx.x;
  const int b  = bh / H_;
  const int h  = bh - b * H_;

  for (int i = tid; i < 64 * 64; i += 128) Mst[i] = 0.0f;
  if (tid < 64) KsS[tid] = 0.0f;
  __syncthreads();

  const size_t ld    = (size_t)H_ * D_;                      // 1024
  const size_t base  = (size_t)b * S_ * ld + (size_t)h * D_; // row-major Q/K/out
  const size_t baset = (size_t)bh * 64 * S_;                 // [b][h][row][s]
  const int rowb = wave * 16;
  const int ro = (lane >> 4) << 3;
  const int cL = lane & 15;

  for (int c0 = 0; c0 < S_; c0 += 64) {
    // ---- async-stage Q, K (row-major) and K^T, V^T (pre-transposed) ----
#pragma unroll
    for (int c = tid; c < 256; c += 128) {
      int r = c >> 2, part = c & 3;
      size_t g  = base + (size_t)(c0 + r) * ld + part * 16;
      size_t gt = baset + (size_t)r * S_ + c0 + part * 16;
      async_copy16(Qs  + r * 64 + part * 16, Qg  + g);
      async_copy16(Ks  + r * 64 + part * 16, Kg  + g);
      async_copy16(KTs + r * 64 + part * 16, KTg + gt);
      async_copy16(VTs + r * 64 + part * 16, VTg + gt);
    }
    async_wait<0>();
    __syncthreads();

    // ---- S = Q @ K^T (wave handles rows rowb..rowb+15, 4 col tiles) ----
    v8f sacc[4] = {};
    for (int dd = 0; dd < 64; dd += 32) {
      v16bf aq = afrag(Qs, 64, rowb, dd, lane);
#pragma unroll
      for (int jt = 0; jt < 4; ++jt) {
        v16bf bk = bfrag(Ks, 64, jt * 16, dd, lane);  // col j, K-dim d
        sacc[jt] = wmma_bf16(aq, bk, sacc[jt]);
      }
    }
    // causal mask + store bf16 scores
#pragma unroll
    for (int jt = 0; jt < 4; ++jt)
#pragma unroll
      for (int r = 0; r < 8; ++r) {
        int i = rowb + r + ro, j = jt * 16 + cL;
        float v = (j <= i) ? sacc[jt][r] : 0.0f;
        Ss[i * 64 + j] = (__bf16)v;
      }
    __syncthreads();

    // ---- den_i = q_i . KsS + rowsum(Ss_i) + EPS ----
    if (tid < 64) {
      float s = 0.0f;
      for (int d = 0; d < 64; ++d) s += (float)Qs[tid * 64 + d] * KsS[d];
      for (int j = 0; j < 64; ++j) s += (float)Ss[tid * 64 + j];
      den[tid] = s + 1e-6f;
    }
    __syncthreads();

    // ---- num = Ss @ V + Q @ KV  (shared accumulator) ----
    v8f nacc[4] = {};
    for (int jj = 0; jj < 64; jj += 32) {
      v16bf as = afrag(Ss, 64, rowb, jj, lane);        // masked scores
#pragma unroll
      for (int et = 0; et < 4; ++et) {
        v16bf bv = bfrag(VTs, 64, et * 16, jj, lane);  // col e, K-dim j
        nacc[et] = wmma_bf16(as, bv, nacc[et]);
      }
    }
    for (int dd = 0; dd < 64; dd += 32) {
      v16bf aq = afrag(Qs, 64, rowb, dd, lane);
#pragma unroll
      for (int et = 0; et < 4; ++et) {
        v16bf bm = bfrag_f32(Mst, 64, et * 16, dd, lane);  // col e, K-dim d
        nacc[et] = wmma_bf16(aq, bm, nacc[et]);
      }
    }
    // ---- out_i = num_i / den_i ----
#pragma unroll
    for (int et = 0; et < 4; ++et)
#pragma unroll
      for (int r = 0; r < 8; ++r) {
        int i = rowb + r + ro, e = et * 16 + cL;
        float v = nacc[et][r] / den[i];
        Og[base + (size_t)(c0 + i) * ld + e] = (__bf16)v;
      }
    __syncthreads();  // all reads of Mst / KsS for this chunk done

    // ---- state update: Mst += V^T @ K  (wave handles rows e in strip) ----
    v8f uacc[4] = {};
    for (int jj = 0; jj < 64; jj += 32) {
      v16bf av = afrag(VTs, 64, rowb, jj, lane);
#pragma unroll
      for (int dt = 0; dt < 4; ++dt) {
        v16bf bk = bfrag(KTs, 64, dt * 16, jj, lane);  // col d, K-dim j
        uacc[dt] = wmma_bf16(av, bk, uacc[dt]);
      }
    }
#pragma unroll
    for (int dt = 0; dt < 4; ++dt)
#pragma unroll
      for (int r = 0; r < 8; ++r) {
        int e = rowb + r + ro, d = dt * 16 + cL;
        Mst[e * 64 + d] += uacc[dt][r];
      }
    if (tid < 64) {
      float s = KsS[tid];
      for (int j = 0; j < 64; ++j) s += (float)KTs[tid * 64 + j];
      KsS[tid] = s;
    }
    __syncthreads();
  }
}

// ---------------------------------------------------------------------------
// Launcher
// ---------------------------------------------------------------------------
extern "C" void kernel_launch(void* const* d_in, const int* in_sizes, int n_in,
                              void* d_out, int out_size, void* d_ws, size_t ws_size,
                              hipStream_t stream) {
  (void)in_sizes; (void)n_in; (void)out_size; (void)ws_size;

  const float* x  = (const float*)d_in[0];
  const float* Wq = (const float*)d_in[1];
  const float* bq = (const float*)d_in[2];
  const float* Wk = (const float*)d_in[3];
  const float* bk = (const float*)d_in[4];
  const float* Wv = (const float*)d_in[5];
  const float* bv = (const float*)d_in[6];
  const float* Wo = (const float*)d_in[7];
  const float* bo = (const float*)d_in[8];
  float* out = (float*)d_out;

  char* ws = (char*)d_ws;
  const size_t SZ = (size_t)BS_ * HID_ * sizeof(__bf16);  // 32 MB
  __bf16* xb  = (__bf16*)(ws);
  __bf16* Qb  = (__bf16*)(ws + 1 * SZ);
  __bf16* Kb  = (__bf16*)(ws + 2 * SZ);
  __bf16* KTb = (__bf16*)(ws + 3 * SZ);   // [b][h][d][s]
  __bf16* VTb = (__bf16*)(ws + 4 * SZ);   // [b][h][e][s]
  __bf16* Ab  = (__bf16*)(ws + 5 * SZ);
  const size_t WSZ = (size_t)HID_ * HID_ * sizeof(__bf16);  // 2 MB
  __bf16* WqT = (__bf16*)(ws + 6 * SZ);
  __bf16* WkT = (__bf16*)(ws + 6 * SZ + 1 * WSZ);
  __bf16* WvT = (__bf16*)(ws + 6 * SZ + 2 * WSZ);
  __bf16* WoT = (__bf16*)(ws + 6 * SZ + 3 * WSZ);

  // prep: x -> bf16, transpose+convert weights
  cvt_f32_bf16_kernel<<<4096, 256, 0, stream>>>(x, xb, BS_ * HID_);
  dim3 tgrid(HID_ / 32, HID_ / 32);
  transpose_cvt_kernel<<<tgrid, 256, 0, stream>>>(Wq, WqT, HID_, HID_);
  transpose_cvt_kernel<<<tgrid, 256, 0, stream>>>(Wk, WkT, HID_, HID_);
  transpose_cvt_kernel<<<tgrid, 256, 0, stream>>>(Wv, WvT, HID_, HID_);
  transpose_cvt_kernel<<<tgrid, 256, 0, stream>>>(Wo, WoT, HID_, HID_);

  // projections (elu+1 fused on Q/K; K,V also stored head-transposed)
  dim3 ggrid(HID_ / 128, BS_ / 128);
  gemm_bf16_kernel<1, 1, 1, 0><<<ggrid, 256, 0, stream>>>(
      xb, WqT, bq, Qb, nullptr, BS_, HID_, HID_);
  gemm_bf16_kernel<1, 1, 1, 1><<<ggrid, 256, 0, stream>>>(
      xb, WkT, bk, Kb, KTb, BS_, HID_, HID_);
  gemm_bf16_kernel<0, 1, 0, 1><<<ggrid, 256, 0, stream>>>(
      xb, WvT, bv, nullptr, VTb, BS_, HID_, HID_);

  // chunked linear-attention scan: one block per (b, h)
  linattn_scan_kernel<<<B_ * H_, 128, 0, stream>>>(Qb, Kb, KTb, VTb, Ab);

  // output projection (f32 out + bias)
  gemm_bf16_kernel<0, 0, 1, 0><<<ggrid, 256, 0, stream>>>(
      Ab, WoT, bo, out, nullptr, BS_, HID_, HID_);
}
